// MoEModel_36756330119410
// MI455X (gfx1250) — matrix operand order repaired and checked by reading, not compiled, not run on hardware
//
#include <hip/hip_runtime.h>
#include <hip/hip_bf16.h>

// Problem constants (match the JAX reference)
#define Bc 32
#define Tc 1024
#define Dc 512
#define Gc 256
#define Ec 8
#define Vc 8192

#define ROWS 64          // T-rows per workgroup (4 row-tiles of 16)
#define COLS 32          // V-columns per chunk  (2 col-groups of 16)

typedef __attribute__((ext_vector_type(16))) __bf16 v16bf;
typedef __attribute__((ext_vector_type(8)))  float  v8f;
typedef __attribute__((ext_vector_type(4)))  float  v4f;

union ABfrag { v16bf v; uint4 q[2]; };

// Packed f32 -> 2x bf16 (RNE) using the gfx1250 packed converter.
__device__ __forceinline__ unsigned cvt2(float lo, float hi) {
    unsigned r;
    asm("v_cvt_pk_bf16_f32 %0, %1, %2" : "=v"(r) : "v"(lo), "v"(hi));
    return r;
}
__device__ __forceinline__ void merge_ms(float& m, float& s, float& t,
                                         float om, float os, float ot) {
    float nm = fmaxf(m, om);
    s = s * __expf(m - nm) + os * __expf(om - nm);
    m = nm;
    t += ot;
}

// ---------------------------------------------------------------------------
// Kernel 1: router.  One block, 256 threads: thread (b,e) computes one logit.
// ---------------------------------------------------------------------------
__global__ void moe_router_kernel(const float* __restrict__ gf,
                                  const float* __restrict__ Wg,
                                  const float* __restrict__ bg,
                                  float* __restrict__ out,
                                  float* __restrict__ ws_acc,
                                  int* __restrict__ ws_assign) {
    __shared__ float sl[Bc][Ec];
    const int tid = threadIdx.x;
    const int b = tid >> 3;
    const int e = tid & 7;
    float s = bg[e];
    for (int g = 0; g < Gc; ++g)
        s = fmaf(gf[b * Gc + g], Wg[g * Ec + e], s);
    sl[b][e] = s;
    __syncthreads();
    if (tid < Bc) {
        float mx = sl[tid][0];
        int am = 0;
        #pragma unroll
        for (int i = 1; i < Ec; ++i)
            if (sl[tid][i] > mx) { mx = sl[tid][i]; am = i; }
        float ex[Ec], den = 0.f;
        #pragma unroll
        for (int i = 0; i < Ec; ++i) { ex[i] = __expf(sl[tid][i] - mx); den += ex[i]; }
        const float inv = 1.f / den;
        #pragma unroll
        for (int i = 0; i < Ec; ++i)
            out[1 + Bc + tid * Ec + i] = ex[i] * inv;   // probs at offset 33
        out[1 + tid] = (float)am;                       // assignments at offset 1
        ws_assign[tid] = am;
        ws_acc[tid] = 0.f;                              // zero NLL accumulator
    }
}

// ---------------------------------------------------------------------------
// Kernel 2: fused expert GEMM + online log-softmax cross-entropy.
// grid = B * (T/64) blocks of 256 threads (8 wave32's).
// Block owns 64 rows of T for one sample b.  Waves tile 4 row-tiles x 2
// col-groups.  A-fragments for the full K=512 are preloaded into registers
// (16 x v16bf = 128 VGPRs) and reused across all 256 column chunks; only the
// W slice streams through a double-buffered LDS tile (1 barrier / K-step).
// ---------------------------------------------------------------------------
__global__ void __launch_bounds__(256)
moe_expert_ce_kernel(const float* __restrict__ x,
                     const float* __restrict__ We,
                     const float* __restrict__ be,
                     const int*   __restrict__ tgt,
                     const int*   __restrict__ ws_assign,
                     float*       __restrict__ ws_acc) {
    __shared__ __align__(16) unsigned short A_lds[ROWS][Dc];       // 64 KB bf16 activations
    __shared__ __align__(16) unsigned short B_lds[2][COLS][32];    // 2x2 KB bf16 W (transposed)
    __shared__ float red_m2[2][ROWS];
    __shared__ float red_s2[2][ROWS];
    __shared__ float red_t2[2][ROWS];
    __shared__ int   tgt_lds[ROWS];
    __shared__ float nll_lds[ROWS];

    const int tid  = threadIdx.x;
    const int lane = tid & 31;
    const int wave = tid >> 5;
    const int blk  = blockIdx.x;
    const int b    = blk >> 4;            // 16 row-blocks per sample
    const int t0   = (blk & 15) * ROWS;

    const int e = ws_assign[b];
    const float* __restrict__ Wexp = We + (size_t)e * Dc * Vc;
    const float* __restrict__ bexp = be + (size_t)e * Vc;

    // ---- stage A: 64 rows x 512 f32 -> bf16 in LDS.  Non-temporal loads:
    // x is streamed exactly once, keep L2 for the resident We working set. ----
    {
        const int row = tid >> 2;
        const int cb  = (tid & 3) * 128;
        const v4f* s4 = (const v4f*)(x + ((size_t)b * Tc + t0 + row) * Dc + cb);
        uint4* dst = (uint4*)&A_lds[row][cb];
        #pragma unroll
        for (int j = 0; j < 32; j += 2) {
            v4f va = __builtin_nontemporal_load(s4 + j);
            v4f vb = __builtin_nontemporal_load(s4 + j + 1);
            uint4 u;
            u.x = cvt2(va.x, va.y);
            u.y = cvt2(va.z, va.w);
            u.z = cvt2(vb.x, vb.y);
            u.w = cvt2(vb.z, vb.w);
            dst[j >> 1] = u;
        }
    }
    if (tid < ROWS) tgt_lds[tid] = tgt[(size_t)b * Tc + t0 + tid];
    __syncthreads();

    // ---- per-lane WMMA fragment geometry (wave32, 16x16x32 bf16 layouts) ----
    const int jcol    = lane & 15;
    const int hi      = (lane >= 16) ? 1 : 0;
    const int rowbase = hi * 8;              // C/D rows in this half-wave
    const int rt      = wave >> 1;           // row-tile 0..3
    const int cg      = wave & 1;            // col-group 0..1
    const int a_row   = rt * 16 + jcol;      // A-matrix row for this lane
    const int a_koff  = hi * 8;              // A K runs: {a_koff.., a_koff+16..}
    const int b_koff  = hi * 8;              // B K runs: {b_koff.., b_koff+16..}
    const int b_col   = cg * 16 + jcol;      // column within the 32-col chunk

    // ---- preload ALL 16 A-fragments (full K=512) into registers ----
    ABfrag afr[16];
    #pragma unroll
    for (int kc = 0; kc < Dc / 32; ++kc) {
        const uint4* ap = (const uint4*)&A_lds[a_row][kc * 32 + a_koff];
        afr[kc].q[0] = ap[0];                // K = base .. base+7
        afr[kc].q[1] = ap[2];                // K = base+16 .. base+23
    }

    int trow[8];
    #pragma unroll
    for (int r = 0; r < 8; ++r) trow[r] = tgt_lds[rt * 16 + rowbase + r];

    float mrun[8], srun[8], tacc[8];
    #pragma unroll
    for (int r = 0; r < 8; ++r) { mrun[r] = -__builtin_inff(); srun[r] = 0.f; tacc[r] = 0.f; }

    // B staging mapping: thread owns column (tid&31) and a 4-deep K strip.
    const int bn  = tid & 31;
    const int bks = (tid >> 5) * 4;

    #pragma unroll 1
    for (int ct = 0; ct < Vc / COLS; ++ct) {             // 256 column chunks
        const int v0 = ct * COLS;
        v8f acc = {};

        // prologue: stage K-step 0 into buffer 0
        {
            const float* wsrc = Wexp + (size_t)bks * Vc + (v0 + bn);
            float f0 = wsrc[0];
            float f1 = wsrc[(size_t)Vc];
            float f2 = wsrc[(size_t)2 * Vc];
            float f3 = wsrc[(size_t)3 * Vc];
            uint2 u;
            u.x = cvt2(f0, f1);
            u.y = cvt2(f2, f3);
            *(uint2*)&B_lds[0][bn][bks] = u;
        }
        __syncthreads();

        #pragma unroll
        for (int kc = 0; kc < Dc / 32; ++kc) {           // 16 K-steps, full unroll
            const int cur = kc & 1;
            float n0 = 0.f, n1 = 0.f, n2 = 0.f, n3 = 0.f;
            if (kc < 15) {                               // prefetch next K-step
                const float* wsrc =
                    Wexp + (size_t)((kc + 1) * 32 + bks) * Vc + (v0 + bn);
                n0 = wsrc[0];
                n1 = wsrc[(size_t)Vc];
                n2 = wsrc[(size_t)2 * Vc];
                n3 = wsrc[(size_t)3 * Vc];
            }
            ABfrag bf;
            const uint4* bp = (const uint4*)&B_lds[cur][b_col][b_koff];
            bf.q[0] = bp[0];
            bf.q[1] = bp[2];
            acc = __builtin_amdgcn_wmma_f32_16x16x32_bf16(
                false, afr[kc].v, false, bf.v, (short)0, acc, false, false);
            if (kc < 15) {                               // fill other buffer
                uint2 u;
                u.x = cvt2(n0, n1);
                u.y = cvt2(n2, n3);
                *(uint2*)&B_lds[1 - cur][bn][bks] = u;
            }
            __syncthreads();
        }

        // ---- epilogue: bias + online softmax + target-logit gather ----
        const int   mycol = v0 + b_col;
        const float bias  = bexp[mycol];
        #pragma unroll
        for (int r = 0; r < 8; ++r) {
            float val = acc[r] + bias;
            float nm  = fmaxf(mrun[r], val);
            srun[r] = srun[r] * __expf(mrun[r] - nm) + __expf(val - nm);
            mrun[r] = nm;
            if (mycol == trow[r]) tacc[r] += val;
        }
    }

    // ---- reduce across the 16 lane-columns sharing each row (shuffles) ----
    #pragma unroll
    for (int r = 0; r < 8; ++r) {
        #pragma unroll
        for (int mask = 1; mask <= 8; mask <<= 1) {
            float om = __shfl_xor(mrun[r], mask);
            float os = __shfl_xor(srun[r], mask);
            float ot = __shfl_xor(tacc[r], mask);
            merge_ms(mrun[r], srun[r], tacc[r], om, os, ot);
        }
        if (jcol == 0) {
            const int row = rt * 16 + rowbase + r;
            red_m2[cg][row] = mrun[r];
            red_s2[cg][row] = srun[r];
            red_t2[cg][row] = tacc[r];
        }
    }
    __syncthreads();

    // ---- combine the two col-groups, per row ----
    if (tid < ROWS) {
        float m0 = red_m2[0][tid], s0 = red_s2[0][tid];
        float m1 = red_m2[1][tid], s1 = red_s2[1][tid];
        float M  = fmaxf(m0, m1);
        float S  = s0 * __expf(m0 - M) + s1 * __expf(m1 - M);
        float Tg = red_t2[0][tid] + red_t2[1][tid];
        nll_lds[tid] = (M + __logf(S)) - Tg;             // logsumexp - target logit
    }
    __syncthreads();
    if (tid == 0) {
        float s = 0.f;
        #pragma unroll
        for (int i = 0; i < ROWS; ++i) s += nll_lds[i];
        atomicAdd(&ws_acc[b], s);
    }
}

// ---------------------------------------------------------------------------
// Kernel 3: combine per-sample NLL sums into per-expert means + total loss.
// ---------------------------------------------------------------------------
__global__ void moe_finalize_kernel(const float* __restrict__ ws_acc,
                                    const int*   __restrict__ ws_assign,
                                    float* __restrict__ out) {
    if (threadIdx.x == 0) {
        float sums[Ec] = {}, cnt[Ec] = {};
        for (int b = 0; b < Bc; ++b) {
            float ce = ws_acc[b] * (1.0f / (float)Tc);
            int e = ws_assign[b];
            sums[e] += ce;
            cnt[e]  += 1.f;
        }
        float loss = 0.f;
        #pragma unroll
        for (int e = 0; e < Ec; ++e)
            if (cnt[e] > 0.f) loss += sums[e] / cnt[e];
        out[0] = loss;
    }
}

extern "C" void kernel_launch(void* const* d_in, const int* in_sizes, int n_in,
                              void* d_out, int out_size, void* d_ws, size_t ws_size,
                              hipStream_t stream) {
    const float* gate = (const float*)d_in[0];   // [B, G]
    const float* x    = (const float*)d_in[1];   // [B, T, D]
    const float* Wg   = (const float*)d_in[2];   // [G, E]
    const float* bg   = (const float*)d_in[3];   // [E]
    const float* We   = (const float*)d_in[4];   // [E, D, V]
    const float* be   = (const float*)d_in[5];   // [E, V]
    const int*   tg   = (const int*)d_in[6];     // [B, T]

    float* out       = (float*)d_out;            // [1 + B + B*E]
    float* ws_acc    = (float*)d_ws;             // [B] per-sample NLL sums
    int*   ws_assign = (int*)((char*)d_ws + 128);

    moe_router_kernel<<<1, 256, 0, stream>>>(gate, Wg, bg, out, ws_acc, ws_assign);
    moe_expert_ce_kernel<<<dim3(Bc * (Tc / ROWS)), dim3(256), 0, stream>>>(
        x, We, be, tg, ws_assign, ws_acc);
    moe_finalize_kernel<<<1, 64, 0, stream>>>(ws_acc, ws_assign, out);
}
